// CrossAttention_29480655519762
// MI455X (gfx1250) — compile-verified, matrix operand order checked by reading, not run on hardware
//
#include <hip/hip_runtime.h>
#include <hip/hip_bf16.h>
#include <hip/hip_fp16.h>
#include <cstddef>
#include <cstdint>

// ---------------- CDNA5 WMMA types ----------------
typedef __attribute__((ext_vector_type(16))) _Float16 v16h;
typedef __attribute__((ext_vector_type(8)))  _Float16 v8h;
typedef __attribute__((ext_vector_type(4)))  _Float16 v4h;
typedef __attribute__((ext_vector_type(8)))  float    v8f;

#define DEVINL __device__ __forceinline__

static constexpr int CDIM  = 1024;   // model dim
static constexpr int NHEAD = 16;
static constexpr int HD    = 64;     // head dim
static constexpr int BATCH = 4;
static constexpr int NQ    = 2048;
static constexpr int NK    = 2048;
static constexpr float SCALE = 0.125f; // 64^-0.5

DEVINL v8f wmma16(v16h a, v16h b, v8f c) {
  // D = A(16x32 f16) * B(32x16 f16) + C(16x16 f32)
  return __builtin_amdgcn_wmma_f32_16x16x32_f16(
      /*neg_a=*/false, a, /*neg_b=*/false, b,
      /*c_mod=*/(short)0, c, /*reuse_a=*/false, /*reuse_b=*/false);
}

// Async DMA, GVS addressing: mem_addr = SGPR64 base + per-lane i32 offset.
// 16 bytes per lane, global -> LDS, tracked by ASYNCcnt. The low 32 bits of
// a generic pointer into __shared__ are the LDS byte address (ISA 10.2:
// flat->LDS truncates to addr[31:0], relative to the wave's LDS base).
DEVINL void async_ld16(void* lds_p, const void* base_u, uint32_t voff) {
  asm volatile("global_load_async_to_lds_b128 %0, %1, %2"
               :: "v"((uint32_t)(uintptr_t)lds_p), "v"(voff),
                  "s"((uint64_t)(uintptr_t)base_u)
               : "memory");
}
DEVINL void wait_async0() { asm volatile("s_wait_asynccnt 0x0" ::: "memory"); }

// Load one 16x32 fp16 A-fragment (or col-major B-fragment) for this lane.
// rowp points at the lane's matrix row (row-major, K contiguous).
// Layout (ISA 7.12.2, wave32): lanes 0-15 hold K = ks+[0..7] and ks+16+[0..7];
// lanes 16-31 hold K = ks+8+[0..7] and ks+24+[0..7].
DEVINL v16h load_frag(const _Float16* rowp, int ks, int half) {
  v8h lo = *(const v8h*)(rowp + ks + half * 8);
  v8h hi = *(const v8h*)(rowp + ks + 16 + half * 8);
  v16h f;
#pragma unroll
  for (int i = 0; i < 8; ++i) { f[i] = lo[i]; f[i + 8] = hi[i]; }
  return f;
}

// ---------------- stage 1: fp32 -> fp16 convert ----------------
__global__ void convert_f16_kernel(const float* __restrict__ src,
                                   _Float16* __restrict__ dst, long n4) {
  long i = (long)blockIdx.x * blockDim.x + threadIdx.x;
  if (i < n4) {
    float4 v = ((const float4*)src)[i];
    v4h o = { (_Float16)v.x, (_Float16)v.y, (_Float16)v.z, (_Float16)v.w };
    ((v4h*)dst)[i] = o;
  }
}

// ---------------- stage 2: weight transpose + convert ----------------
// W[k][n] fp32 -> Wt[n][k] fp16  (so WMMA B-fragment reads are K-contiguous)
__global__ __launch_bounds__(1024)
void convert_wT_kernel(const float* __restrict__ W, _Float16* __restrict__ Wt,
                       int K, int N) {
  __shared__ float tile[32][33];
  int tx = threadIdx.x, ty = threadIdx.y;
  int k = blockIdx.y * 32 + ty;
  int n = blockIdx.x * 32 + tx;
  tile[ty][tx] = W[(size_t)k * N + n];
  __syncthreads();
  int n2 = blockIdx.x * 32 + ty;
  int k2 = blockIdx.y * 32 + tx;
  Wt[(size_t)n2 * K + k2] = (_Float16)tile[tx][ty];
}

// ---------------- stage 3/5: WMMA GEMM (async double-buffered) ------------
// C[M][N] = A[M][K] @ B[K][N] + bias,  B given pre-transposed as Bt[N][K].
// 256 threads = 8 waves; block tile 128x128; wave tile 32x64 (2x4 WMMA tiles).
// K-slabs DMA'd into LDS (GVS async, SGPR base + small per-lane offset) with
// ASYNCcnt double buffering: one wait + one barrier per 64-wide K-step; the
// copy of slab i+1 overlaps WMMA on slab i.
template <typename OUT>
__global__ __launch_bounds__(256)
void gemm_wmma_kernel(const _Float16* __restrict__ A,
                      const _Float16* __restrict__ Bt,
                      const float* __restrict__ bias,
                      OUT* __restrict__ C, int Mdim, int Ndim, int Kdim) {
  __shared__ __align__(16) _Float16 As[2][128][72];
  __shared__ __align__(16) _Float16 Bs[2][128][72];

  const int tid  = threadIdx.x;
  const int lane = tid & 31;
  const int wave = tid >> 5;
  const int half = lane >> 4;
  const int lm   = lane & 15;
  const int mBase = blockIdx.y * 128;
  const int nBase = blockIdx.x * 128;
  const int wr = wave & 3;   // wave row -> m offset wr*32
  const int wc = wave >> 2;  // wave col -> n offset wc*64

  // uniform slab bases (SGPRs); per-lane part is a 32-bit byte offset
  const _Float16* aBase0 = A  + (size_t)mBase * Kdim;
  const _Float16* bBase0 = Bt + (size_t)nBase * Kdim;

  auto issue_slab = [&](int buf, int kb) {
    const _Float16* aB = aBase0 + kb;
    const _Float16* bB = bBase0 + kb;
#pragma unroll
    for (int s = tid; s < 1024; s += 256) {
      int row = s >> 3, ch = (s & 7) * 8;
      uint32_t voff = (uint32_t)(row * Kdim + ch) * 2u;  // bytes
      async_ld16(&As[buf][row][ch], aB, voff);
      async_ld16(&Bs[buf][row][ch], bB, voff);
    }
  };

  v8f acc[2][4];
  const v8f zf = {0.f, 0.f, 0.f, 0.f, 0.f, 0.f, 0.f, 0.f};
#pragma unroll
  for (int mt = 0; mt < 2; ++mt)
#pragma unroll
    for (int nt = 0; nt < 4; ++nt) acc[mt][nt] = zf;

  issue_slab(0, 0);

  for (int kb = 0; kb < Kdim; kb += 64) {
    const int cur = (kb >> 6) & 1;
    wait_async0();      // my slab-DMAs into As/Bs[cur] are complete
    __syncthreads();    // everyone's are; everyone done reading As/Bs[cur^1]
    if (kb + 64 < Kdim) issue_slab(cur ^ 1, kb + 64);

#pragma unroll
    for (int ks = 0; ks < 64; ks += 32) {
      v16h af[2], bf[4];
#pragma unroll
      for (int mt = 0; mt < 2; ++mt)
        af[mt] = load_frag(&As[cur][wr * 32 + mt * 16 + lm][0], ks, half);
#pragma unroll
      for (int nt = 0; nt < 4; ++nt)
        bf[nt] = load_frag(&Bs[cur][wc * 64 + nt * 16 + lm][0], ks, half);
#pragma unroll
      for (int mt = 0; mt < 2; ++mt)
#pragma unroll
        for (int nt = 0; nt < 4; ++nt)
          acc[mt][nt] = wmma16(af[mt], bf[nt], acc[mt][nt]);
    }
  }

  // epilogue: C-layout = lane (half,n) holds rows 8*half+r in element r
#pragma unroll
  for (int mt = 0; mt < 2; ++mt)
#pragma unroll
    for (int nt = 0; nt < 4; ++nt) {
      int ncol = nBase + wc * 64 + nt * 16 + lm;
      float bv = bias ? bias[ncol] : 0.f;
#pragma unroll
      for (int r = 0; r < 8; ++r) {
        int mrow = mBase + wr * 32 + mt * 16 + half * 8 + r;
        C[(size_t)mrow * Ndim + ncol] = (OUT)(acc[mt][nt][r] + bv);
      }
    }
}

// ---------------- stage 4: flash attention (WMMA, online softmax) ----------
// grid = (NQ/128, BATCH*NHEAD), block = 256 (8 waves), wave owns 16 q rows.
// K tiles: async DMA to LDS (GVS form), double buffered. V tiles: regular
// load + transposed LDS store (so the P*V B-fragment is key-contiguous).
// Single barrier per 64-key tile.
__global__ __launch_bounds__(256)
void attn_wmma_kernel(const _Float16* __restrict__ Qp,
                      const _Float16* __restrict__ Kp,
                      const _Float16* __restrict__ Vp,
                      _Float16* __restrict__ AO) {
  __shared__ __align__(16) _Float16 Ks[2][64][72];     // K tile, row-major
  __shared__ __align__(16) _Float16 Vst[2][64][72];    // V tile, [d][key]
  __shared__ __align__(16) _Float16 Ps[8][16][72];     // per-wave P strip

  const int tid  = threadIdx.x;
  const int lane = tid & 31;
  const int wave = tid >> 5;
  const int half = lane >> 4;
  const int lm   = lane & 15;
  const int bh = blockIdx.y;
  const int b  = bh >> 4;
  const int h  = bh & 15;
  const int q0 = blockIdx.x * 128 + wave * 16;

  const _Float16* kBase0 = Kp + ((size_t)b * NK) * CDIM + h * HD;
  const _Float16* vBase0 = Vp + ((size_t)b * NK) * CDIM + h * HD;

  auto load_kv = [&](int buf, int kt) {
    const _Float16* kB = kBase0 + (size_t)kt * CDIM;
    const _Float16* vB = vBase0 + (size_t)kt * CDIM;
#pragma unroll
    for (int s = tid; s < 512; s += 256) {
      int row = s >> 3, ch = (s & 7) * 8;
      uint32_t voff = (uint32_t)(row * CDIM + ch) * 2u;  // bytes
      async_ld16(&Ks[buf][row][ch], kB, voff);           // ASYNCcnt path
      v8h vv = *(const v8h*)(vB + row * CDIM + ch);
#pragma unroll
      for (int i = 0; i < 8; ++i) Vst[buf][ch + i][row] = vv[i];
    }
  };

  // Q fragments for this wave's 16 rows, kept in registers all kernel.
  const _Float16* qrow = Qp + ((size_t)(b * NQ + q0 + lm)) * CDIM + h * HD;
  v16h qf[2];
  qf[0] = load_frag(qrow, 0, half);
  qf[1] = load_frag(qrow, 32, half);

  float mrun[8], lrun[8];
  v8f o[4];
  const v8f zf = {0.f, 0.f, 0.f, 0.f, 0.f, 0.f, 0.f, 0.f};
#pragma unroll
  for (int r = 0; r < 8; ++r) { mrun[r] = -1e30f; lrun[r] = 0.f; }
#pragma unroll
  for (int nt = 0; nt < 4; ++nt) o[nt] = zf;

  load_kv(0, 0);

  for (int kt = 0; kt < NK; kt += 64) {
    const int cur = (kt >> 6) & 1;
    wait_async0();      // my K-DMAs into Ks[cur] done (dscnt waited at barrier)
    __syncthreads();    // tile cur fully visible; tile cur^1 free for reuse
    if (kt + 64 < NK) load_kv(cur ^ 1, kt + 64);

    // S = Q * K^T  (K rows are B-matrix columns, K-contiguous in Ks)
    v8f sacc[4];
#pragma unroll
    for (int nt = 0; nt < 4; ++nt) sacc[nt] = zf;
#pragma unroll
    for (int ks = 0; ks < 64; ks += 32) {
#pragma unroll
      for (int nt = 0; nt < 4; ++nt) {
        v16h bf = load_frag(&Ks[cur][nt * 16 + lm][0], ks, half);
        sacc[nt] = wmma16(qf[ks >> 5], bf, sacc[nt]);
      }
    }

    // online softmax on the C-layout (row 8*half+r lives in element r,
    // spread over the 16 lanes of this half)
    float pv[4][8];
#pragma unroll
    for (int r = 0; r < 8; ++r) {
      float mx = fmaxf(fmaxf(sacc[0][r], sacc[1][r]),
                       fmaxf(sacc[2][r], sacc[3][r]));
#pragma unroll
      for (int m = 1; m < 16; m <<= 1) mx = fmaxf(mx, __shfl_xor(mx, m, 32));
      float nm = fmaxf(mrun[r], mx * SCALE);
      float alpha = __expf(mrun[r] - nm);
      float rs = 0.f;
#pragma unroll
      for (int nt = 0; nt < 4; ++nt) {
        float p = __expf(sacc[nt][r] * SCALE - nm);
        pv[nt][r] = p;
        rs += p;
      }
#pragma unroll
      for (int m = 1; m < 16; m <<= 1) rs += __shfl_xor(rs, m, 32);
      lrun[r] = lrun[r] * alpha + rs;
      mrun[r] = nm;
#pragma unroll
      for (int nt = 0; nt < 4; ++nt) o[nt][r] *= alpha;
    }

    // P -> per-wave LDS strip (fp16), then re-read in A-fragment layout.
#pragma unroll
    for (int nt = 0; nt < 4; ++nt)
#pragma unroll
      for (int r = 0; r < 8; ++r)
        Ps[wave][half * 8 + r][nt * 16 + lm] = (_Float16)pv[nt][r];
    // same-wave LDS RAW: hardware keeps DS ops in order; wait out the count.
    asm volatile("s_wait_dscnt 0x0" ::: "memory");

    // O += P * V  (V^T columns are key-contiguous in Vst)
#pragma unroll
    for (int ks = 0; ks < 64; ks += 32) {
      v16h pf = load_frag(&Ps[wave][lm][0], ks, half);
#pragma unroll
      for (int nt = 0; nt < 4; ++nt) {
        v16h vf = load_frag(&Vst[cur][nt * 16 + lm][0], ks, half);
        o[nt] = wmma16(pf, vf, o[nt]);
      }
    }
  }

  // normalize + store fp16 to [B][Nq][C] with head h at cols h*64..h*64+63
#pragma unroll
  for (int r = 0; r < 8; ++r) {
    float inv = 1.f / lrun[r];
    int q = q0 + half * 8 + r;
    size_t rowo = ((size_t)(b * NQ + q)) * CDIM + h * HD;
#pragma unroll
    for (int nt = 0; nt < 4; ++nt)
      AO[rowo + nt * 16 + lm] = (_Float16)(o[nt][r] * inv);
  }
}

// ---------------- host-side orchestration ----------------
extern "C" void kernel_launch(void* const* d_in, const int* in_sizes, int n_in,
                              void* d_out, int out_size, void* d_ws, size_t ws_size,
                              hipStream_t stream) {
  (void)in_sizes; (void)n_in; (void)out_size; (void)ws_size;
  const float* query = (const float*)d_in[0];
  const float* key   = (const float*)d_in[1];
  const float* value = (const float*)d_in[2];
  const float* Wq = (const float*)d_in[5];
  const float* bq = (const float*)d_in[6];
  const float* Wk = (const float*)d_in[7];
  const float* bk = (const float*)d_in[8];
  const float* Wv = (const float*)d_in[9];
  const float* bv = (const float*)d_in[10];
  const float* Wo = (const float*)d_in[11];
  const float* bo = (const float*)d_in[12];
  float* out = (float*)d_out;

  const size_t MV = (size_t)BATCH * NQ * CDIM;   // 8M elements
  const size_t WE = (size_t)CDIM * CDIM;         // 1M elements
  _Float16* ws  = (_Float16*)d_ws;
  _Float16* Xq  = ws;                 // fp16 inputs
  _Float16* Xk  = Xq  + MV;
  _Float16* Xv  = Xk  + MV;
  _Float16* WqT = Xv  + MV;           // fp16 transposed weights
  _Float16* WkT = WqT + WE;
  _Float16* WvT = WkT + WE;
  _Float16* WoT = WvT + WE;
  _Float16* Qp  = WoT + WE;           // projected q/k/v (fp16)
  _Float16* Kp  = Qp  + MV;
  _Float16* Vp  = Kp  + MV;
  _Float16* AO  = Vp  + MV;           // attention output (fp16)

  // stage 1: convert inputs
  {
    long n4 = (long)(MV / 4);
    int blk = 256, grd = (int)((n4 + blk - 1) / blk);
    convert_f16_kernel<<<grd, blk, 0, stream>>>(query, Xq, n4);
    convert_f16_kernel<<<grd, blk, 0, stream>>>(key,   Xk, n4);
    convert_f16_kernel<<<grd, blk, 0, stream>>>(value, Xv, n4);
  }
  // stage 2: transpose + convert weights
  {
    dim3 g(CDIM / 32, CDIM / 32), t(32, 32);
    convert_wT_kernel<<<g, t, 0, stream>>>(Wq, WqT, CDIM, CDIM);
    convert_wT_kernel<<<g, t, 0, stream>>>(Wk, WkT, CDIM, CDIM);
    convert_wT_kernel<<<g, t, 0, stream>>>(Wv, WvT, CDIM, CDIM);
    convert_wT_kernel<<<g, t, 0, stream>>>(Wo, WoT, CDIM, CDIM);
  }
  // stage 3: q/k/v projections
  {
    dim3 g(CDIM / 128, (BATCH * NQ) / 128);
    gemm_wmma_kernel<_Float16><<<g, 256, 0, stream>>>(Xq, WqT, bq, Qp,
                                                      BATCH * NQ, CDIM, CDIM);
    gemm_wmma_kernel<_Float16><<<g, 256, 0, stream>>>(Xk, WkT, bk, Kp,
                                                      BATCH * NK, CDIM, CDIM);
    gemm_wmma_kernel<_Float16><<<g, 256, 0, stream>>>(Xv, WvT, bv, Vp,
                                                      BATCH * NK, CDIM, CDIM);
  }
  // stage 4: attention
  {
    dim3 g(NQ / 128, BATCH * NHEAD);
    attn_wmma_kernel<<<g, 256, 0, stream>>>(Qp, Kp, Vp, AO);
  }
  // stage 5: output projection (fp32 out + bias)
  {
    dim3 g(CDIM / 128, (BATCH * NQ) / 128);
    gemm_wmma_kernel<float><<<g, 256, 0, stream>>>(AO, WoT, bo, out,
                                                   BATCH * NQ, CDIM, CDIM);
  }
}
